// MyMultiQueryAttention_46643344835065
// MI455X (gfx1250) — matrix operand order repaired and checked by reading, not compiled, not run on hardware
//
#include <hip/hip_runtime.h>
#include <hip/hip_bf16.h>
#include <math.h>

// Problem constants (B, S, H, NH, D) from the reference.
static constexpr int cB = 2, cS = 2048, cH = 4096, cNH = 32, cD = 128;

typedef __attribute__((ext_vector_type(16))) __bf16   v16bf;
typedef __attribute__((ext_vector_type(8)))  float    v8f;
typedef __attribute__((ext_vector_type(4)))  unsigned uint4v;
typedef __attribute__((ext_vector_type(4)))  float    float4v;

// ---- bf16 helpers (RNE via bit math; used only in one-shot converts and
// ---- epilogues, never in WMMA inner loops) -------------------------------
static __device__ __forceinline__ unsigned bfbits(float f) {
  unsigned u = __float_as_uint(f);
  return (u + 0x7FFFu + ((u >> 16) & 1u)) >> 16;
}
static __device__ __forceinline__ unsigned pack_bf2(float lo, float hi) {
  return (bfbits(hi) << 16) | (bfbits(lo) & 0xFFFFu);
}
static __device__ __forceinline__ __bf16 f2bf(float f) {
  unsigned short h = (unsigned short)bfbits(f);
  __bf16 out;
  __builtin_memcpy(&out, &h, sizeof(out));
  return out;
}

// ---- One-shot streaming f32 -> bf16 convert (8 elems/thread, b128 I/O) ---
__global__ __launch_bounds__(256)
void cvt_f32_to_bf16(const float* __restrict__ in, unsigned short* __restrict__ out,
                     long long n) {
  long long i = ((long long)blockIdx.x * 256 + threadIdx.x) * 8;
  if (i >= n) return;
  float4v f0 = *(const float4v*)(in + i);
  float4v f1 = *(const float4v*)(in + i + 4);
  uint4v o;
  o.x = pack_bf2(f0.x, f0.y);
  o.y = pack_bf2(f0.z, f0.w);
  o.z = pack_bf2(f1.x, f1.y);
  o.w = pack_bf2(f1.z, f1.w);
  *(uint4v*)(out + i) = o;
}

// ---- WMMA fragment loaders from bf16 memory (pure 2x b128 per fragment) --
// A-matrix 16x32: lane half h reads k = h*8+0..7 and 16+h*8+0..7, both
// contiguous runs of 8 bf16 (16B each) -> two 128-bit loads, zero VALU.
static __device__ __forceinline__ v16bf load_a_frag_bf(const __bf16* __restrict__ A,
                                                       int lda) {
  const int lane = threadIdx.x & 31;
  const __bf16* row = A + (size_t)(lane & 15) * lda + ((lane >> 4) << 3);
  union { v16bf v; uint4v q[2]; } u;
  u.q[0] = *(const uint4v*)(row);
  u.q[1] = *(const uint4v*)(row + 16);
  return u.v;
}

// B-matrix 32x16 where B[k][n] = W[n][k], W row-major NxK bf16: lane reads
// k = h*8+0..7 and 16+h*8+0..7 (N=lane%16, h=lane/16) -> two 128-bit loads.
static __device__ __forceinline__ v16bf load_b_frag_bf(const __bf16* __restrict__ W,
                                                       int ldw) {
  const int lane = threadIdx.x & 31;
  const __bf16* p = W + (size_t)(lane & 15) * ldw + ((lane >> 4) << 3);
  union { v16bf v; uint4v q[2]; } u;
  u.q[0] = *(const uint4v*)(p);
  u.q[1] = *(const uint4v*)(p + 16);
  return u.v;
}

// ---- Generic C = X * W^T + bias GEMM, bf16 WMMA, f32 accumulate ----------
// OMODE: 0 = f32 row-major out, 1 = bf16 row-major out,
//        2 = bf16 transposed-per-batch out (B, N, S)  [for V^T]
// Block 128 threads = 4 waves; block tile 128x64; wave tile 32x64.
// Software-pipelined K-loop: the 12 b128 fragment loads for step k+32 are
// issued BEFORE the 8 WMMAs of step k, so WMMA execution overlaps L2/HBM
// latency (the s_wait lands at the register hand-off after the WMMAs).
// MI455X's 192MB L2 caches the whole 33MB bf16 weight matrix, so B-fragment
// re-reads are L2 hits; the streamed operand (X) additionally gets
// global_prefetch a few K-steps ahead.
template <int OMODE>
__global__ __launch_bounds__(128)
void gemm_xwt_bias_wmma(const __bf16* __restrict__ X, const __bf16* __restrict__ W,
                        const float* __restrict__ bias, void* __restrict__ Cout,
                        int M, int N, int K) {
  const int lane = threadIdx.x & 31;
  const int wave = threadIdx.x >> 5;
  const int m0 = blockIdx.y * 128 + wave * 32;
  const int n0 = blockIdx.x * 64;

  v8f acc[2][4] = {};

  // Prologue: fragments for k0 = 0.
  v16bf a0 = load_a_frag_bf(X + (size_t)m0 * K, K);
  v16bf a1 = load_a_frag_bf(X + (size_t)(m0 + 16) * K, K);
  v16bf bfrag[4];
#pragma unroll
  for (int t = 0; t < 4; ++t)
    bfrag[t] = load_b_frag_bf(W + (size_t)(n0 + t * 16) * K, K);

  for (int k0 = 32; k0 < K; k0 += 32) {
    // Prefetch the streamed operand further ahead (global_prefetch_b8).
    if (k0 + 128 < K) {
      __builtin_prefetch(X + (size_t)m0 * K + k0 + 128, 0, 1);
      __builtin_prefetch(X + (size_t)(m0 + 16) * K + k0 + 128, 0, 1);
    }
    // Issue next iteration's loads first...
    const v16bf na0 = load_a_frag_bf(X + (size_t)m0 * K + k0, K);
    const v16bf na1 = load_a_frag_bf(X + (size_t)(m0 + 16) * K + k0, K);
    v16bf nb[4];
#pragma unroll
    for (int t = 0; t < 4; ++t)
      nb[t] = load_b_frag_bf(W + (size_t)(n0 + t * 16) * K + k0, K);
    // ...then compute on the previous iteration's fragments (no wait needed).
#pragma unroll
    for (int t = 0; t < 4; ++t) {
      acc[0][t] = __builtin_amdgcn_wmma_f32_16x16x32_bf16(
          false, a0, false, bfrag[t], (short)0, acc[0][t], false, false);
      acc[1][t] = __builtin_amdgcn_wmma_f32_16x16x32_bf16(
          false, a1, false, bfrag[t], (short)0, acc[1][t], false, false);
    }
    // Register hand-off (this is where the load wait lands).
    a0 = na0;
    a1 = na1;
#pragma unroll
    for (int t = 0; t < 4; ++t) bfrag[t] = nb[t];
  }
  // Epilogue WMMAs for the last K-step.
#pragma unroll
  for (int t = 0; t < 4; ++t) {
    acc[0][t] = __builtin_amdgcn_wmma_f32_16x16x32_bf16(
        false, a0, false, bfrag[t], (short)0, acc[0][t], false, false);
    acc[1][t] = __builtin_amdgcn_wmma_f32_16x16x32_bf16(
        false, a1, false, bfrag[t], (short)0, acc[1][t], false, false);
  }

  const int half = lane >> 4;
  const int n = lane & 15;
#pragma unroll
  for (int s = 0; s < 2; ++s) {
#pragma unroll
    for (int t = 0; t < 4; ++t) {
      const int nc = n0 + t * 16 + n;
      const float bv = bias[nc];
#pragma unroll
      for (int r = 0; r < 8; ++r) {
        const int row = m0 + s * 16 + r + 8 * half;
        const float val = acc[s][t][r] + bv;
        if (OMODE == 0) {
          ((float*)Cout)[(size_t)row * N + nc] = val;
        } else if (OMODE == 1) {
          ((__bf16*)Cout)[(size_t)row * N + nc] = f2bf(val);
        } else {
          const int bb = row / cS;
          const int ss = row % cS;
          ((__bf16*)Cout)[((size_t)bb * N + nc) * cS + ss] = f2bf(val);
        }
      }
    }
  }
}

// ---- Flash MQA attention: one wave per (b, h, 16-query tile) -------------
// Q/K/V^T are pre-converted bf16, so every fragment load is 2x b128.
// Instruction-level latency hiding:
//   * all 8 K-fragments are loaded as one batch before the 8 score WMMAs
//   * the 8 V-fragments are issued BEFORE the softmax, so exp/shfl/LDS work
//     covers their flight time; the wait lands at the P@V WMMAs.
// Raw (unscaled) scores are tracked; 1/sqrt(D) is folded into the exponent.
// Output stored bf16, pre-scrambled as (b, nh, d, s) so the reference's
// swapaxes+reshape becomes a contiguous row-major GEMM input.
__global__ __launch_bounds__(32)
void mqa_flash_wmma(const __bf16* __restrict__ Q,   // (B*S, H) bf16
                    const __bf16* __restrict__ Kc,  // (B, S, D) bf16
                    const __bf16* __restrict__ Vt,  // (B, D, S) bf16
                    __bf16* __restrict__ O) {       // (B, NH, D, S) bf16
  __shared__ float  sm_m[16];
  __shared__ float  sm_l[16];
  __shared__ float  sm_alpha[16];
  __shared__ __bf16 sm_p[16][32];

  const int lane = threadIdx.x & 31;
  const int half = lane >> 4;
  const int q0 = blockIdx.x * 16;
  const int h  = blockIdx.y;
  const int b  = blockIdx.z;

  const float scale = 0.08838834764831845f;  // 1/sqrt(128)

  const __bf16* qbase = Q + (size_t)(b * cS + q0) * cH + h * cD;
  v16bf qa[4];
#pragma unroll
  for (int kk = 0; kk < 4; ++kk) qa[kk] = load_a_frag_bf(qbase + kk * 32, cH);

  v8f oacc[8] = {};
  if (lane < 16) { sm_m[lane] = -1e30f; sm_l[lane] = 0.0f; }
  __syncthreads();

  const __bf16* Kb = Kc + (size_t)b * cS * cD;
  const __bf16* Vb = Vt + (size_t)b * cD * cS;

  for (int kb = 0; kb < cS; kb += 32) {
    // Batch-load all 8 K-fragments (16x b128 in one clause), then compute
    // the two 16x16 score tiles; waits drain progressively.
    v16bf bk[8];
#pragma unroll
    for (int t = 0; t < 2; ++t)
#pragma unroll
      for (int kk = 0; kk < 4; ++kk)
        bk[t * 4 + kk] =
            load_b_frag_bf(Kb + (size_t)(kb + t * 16) * cD + kk * 32, cD);
    v8f st[2] = {};
#pragma unroll
    for (int t = 0; t < 2; ++t)
#pragma unroll
      for (int kk = 0; kk < 4; ++kk)
        st[t] = __builtin_amdgcn_wmma_f32_16x16x32_bf16(
            false, qa[kk], false, bk[t * 4 + kk], (short)0, st[t], false, false);

    // Issue the 8 V-fragment loads now; the whole softmax below overlaps
    // their latency (they're first consumed at the P@V WMMAs).
    v16bf bv[8];
#pragma unroll
    for (int dt = 0; dt < 8; ++dt)
      bv[dt] = load_b_frag_bf(Vb + (size_t)(dt * 16) * cS + kb, cS);

    // Row max over the 32 keys (cross-lane reduce within 16-lane groups).
    float rmax[8];
#pragma unroll
    for (int r = 0; r < 8; ++r) {
      float v = fmaxf(st[0][r], st[1][r]);
#pragma unroll
      for (int off = 8; off > 0; off >>= 1) v = fmaxf(v, __shfl_xor(v, off, 32));
      rmax[r] = v;
    }
    // Lanes 0 and 16 update running max + correction factor for their 8 rows.
    if ((lane & 15) == 0) {
#pragma unroll
      for (int r = 0; r < 8; ++r) {
        const int row = r + 8 * half;
        const float mo = sm_m[row];
        const float mn = fmaxf(mo, rmax[r]);
        sm_m[row] = mn;
        sm_alpha[row] = __expf((mo - mn) * scale);
      }
    }
    __syncthreads();
    float mnew[8], alpha[8];
#pragma unroll
    for (int r = 0; r < 8; ++r) {
      mnew[r]  = sm_m[r + 8 * half];
      alpha[r] = sm_alpha[r + 8 * half];
    }
    // Probabilities (scale folded into exponent) + row sums.
    float p0[8], p1[8], rsum[8];
#pragma unroll
    for (int r = 0; r < 8; ++r) {
      p0[r] = __expf((st[0][r] - mnew[r]) * scale);
      p1[r] = __expf((st[1][r] - mnew[r]) * scale);
      float v = p0[r] + p1[r];
#pragma unroll
      for (int off = 8; off > 0; off >>= 1) v += __shfl_xor(v, off, 32);
      rsum[r] = v;
    }
    if ((lane & 15) == 0) {
#pragma unroll
      for (int r = 0; r < 8; ++r) {
        const int row = r + 8 * half;
        sm_l[row] = sm_l[row] * alpha[r] + rsum[r];
      }
    }
    // Rescale running output accumulator (v_pk_mul_f32).
#pragma unroll
    for (int dt = 0; dt < 8; ++dt)
#pragma unroll
      for (int r = 0; r < 8; ++r) oacc[dt][r] *= alpha[r];
    // Transpose P (C/D layout -> A layout) through LDS, as bf16.
    {
      const int n = lane & 15;
#pragma unroll
      for (int r = 0; r < 8; ++r) {
        sm_p[r + 8 * half][n]      = f2bf(p0[r]);
        sm_p[r + 8 * half][16 + n] = f2bf(p1[r]);
      }
    }
    __syncthreads();
    // Read P back in A-fragment layout: two ds_load_b128 per lane.
    const v16bf pa = load_a_frag_bf(&sm_p[0][0], 32);
    // P @ V^T-tiles into the 16x128 accumulator (V already in flight).
#pragma unroll
    for (int dt = 0; dt < 8; ++dt) {
      oacc[dt] = __builtin_amdgcn_wmma_f32_16x16x32_bf16(
          false, pa, false, bv[dt], (short)0, oacc[dt], false, false);
    }
    __syncthreads();
  }

  // Normalize and store transposed per head: O[b][h][d][s], bf16.
  float linv[8];
#pragma unroll
  for (int r = 0; r < 8; ++r) linv[r] = 1.0f / sm_l[r + 8 * half];
  __bf16* obase = O + (size_t)((b * cNH + h) * cD) * cS;
  const int n = lane & 15;
#pragma unroll
  for (int dt = 0; dt < 8; ++dt) {
    const int d = dt * 16 + n;
#pragma unroll
    for (int r = 0; r < 8; ++r) {
      obase[(size_t)d * cS + (q0 + r + 8 * half)] = f2bf(oacc[dt][r] * linv[r]);
    }
  }
}

extern "C" void kernel_launch(void* const* d_in, const int* in_sizes, int n_in,
                              void* d_out, int out_size, void* d_ws, size_t ws_size,
                              hipStream_t stream) {
  const float* X  = (const float*)d_in[0];  // hidden_state (B,S,H)
  const float* Wq = (const float*)d_in[1];
  const float* bq = (const float*)d_in[2];
  const float* Wk = (const float*)d_in[3];
  const float* bk = (const float*)d_in[4];
  const float* Wv = (const float*)d_in[5];
  const float* bv = (const float*)d_in[6];
  const float* Wo = (const float*)d_in[7];
  const float* bo = (const float*)d_in[8];
  float* out = (float*)d_out;

  const int M = cB * cS;  // 4096 rows
  const long long nXH = (long long)M * cH;   // 16.78M
  const long long nWB = (long long)cH * cH;  // 16.78M
  const long long nWS = (long long)cD * cH;  // 0.52M
  const long long nKV = (long long)cB * cS * cD;

  // Workspace carve (all bf16 / ushort): X | Wq | Wk | Wv | Wo | Q | K | Vt | X2
  unsigned short* w = (unsigned short*)d_ws;
  unsigned short* Xb  = w;            w += nXH;
  unsigned short* Wqb = w;            w += nWB;
  unsigned short* Wkb = w;            w += nWS;
  unsigned short* Wvb = w;            w += nWS;
  unsigned short* Wob = w;            w += nWB;
  unsigned short* Qb  = w;            w += nXH;
  unsigned short* Kbf = w;            w += nKV;
  unsigned short* Vtb = w;            w += nKV;
  unsigned short* X2b = w;            w += nXH;

  auto cvt = [&](const float* src, unsigned short* dst, long long n) {
    const long long blocks = (n / 8 + 255) / 256;
    cvt_f32_to_bf16<<<dim3((unsigned)blocks), dim3(256), 0, stream>>>(src, dst, n);
  };
  // One-shot conversions (streaming, bandwidth-bound).
  cvt(X, Xb, nXH);
  cvt(Wq, Wqb, nWB);
  cvt(Wk, Wkb, nWS);
  cvt(Wv, Wvb, nWS);
  cvt(Wo, Wob, nWB);

  const dim3 blk(128);
  // Q projection -> bf16 row-major (M, H)
  gemm_xwt_bias_wmma<1><<<dim3(cH / 64, M / 128), blk, 0, stream>>>(
      (const __bf16*)Xb, (const __bf16*)Wqb, bq, Qb, M, cH, cH);
  // K projection -> bf16 row-major (M, D)
  gemm_xwt_bias_wmma<1><<<dim3(cD / 64, M / 128), blk, 0, stream>>>(
      (const __bf16*)Xb, (const __bf16*)Wkb, bk, Kbf, M, cD, cH);
  // V projection -> bf16 transposed per batch (B, D, S)
  gemm_xwt_bias_wmma<2><<<dim3(cD / 64, M / 128), blk, 0, stream>>>(
      (const __bf16*)Xb, (const __bf16*)Wvb, bv, Vtb, M, cD, cH);
  // Flash MQA attention -> pre-scrambled bf16 (b, nh, d, s)
  mqa_flash_wmma<<<dim3(cS / 16, cNH, cB), dim3(32), 0, stream>>>(
      (const __bf16*)Qb, (const __bf16*)Kbf, (const __bf16*)Vtb, (__bf16*)X2b);
  // Output projection on the scrambled layout -> f32 out
  gemm_xwt_bias_wmma<0><<<dim3(cH / 64, M / 128), blk, 0, stream>>>(
      (const __bf16*)X2b, (const __bf16*)Wob, bo, out, M, cH, cH);
}
